// PointTransformer_74929999446609
// MI455X (gfx1250) — matrix operand order repaired and checked by reading, not compiled, not run on hardware
//
#include <hip/hip_runtime.h>
#include <hip/hip_bf16.h>

typedef _Float16 half_t;
typedef __attribute__((ext_vector_type(16))) _Float16 v16h;
typedef __attribute__((ext_vector_type(8)))  _Float16 v8h;
typedef __attribute__((ext_vector_type(8)))  float    v8f;

#define BB  2
#define NN  4096
#define KK  16
#define NKT (NN * KK)   // 65536

// ---------------------------------------------------------------------------
// WMMA GEMM, channel-last activations. Each wave owns a 16(M) x NT*16(N)
// super-tile: one shared A fragment, NT B fragments, NT independent
// v_wmma_f32_16x16x32_f16 per 32-K step (overlaps load clauses).
// Epilogue: +bias -> BN(gamma*rsqrt(1+eps), beta) -> +residual -> ReLU.
// Y f16 channel-last (one b128 store per lane per tile); Yf optional f32
// channel-first (reference output layout).
// ---------------------------------------------------------------------------
template <int NT>
__global__ __launch_bounds__(256) void pt_gemm_wmma(
    const half_t* __restrict__ W, int Cinp,
    const half_t* __restrict__ X, long xS, int xLd, int M,
    half_t* __restrict__ Y, long yS, int yLd,
    const float* __restrict__ bias, const float* __restrict__ gamma,
    const float* __restrict__ beta,
    const half_t* __restrict__ res, long resS, int resLd,
    int relu, float* __restrict__ Yf, long yfS)
{
  const int lane = threadIdx.x & 31;
  const int wave = threadIdx.x >> 5;
  const int tN   = blockIdx.x * 8 + wave;   // NT*16-column super-tile
  const int tM   = blockIdx.y;
  const int b    = blockIdx.z;
  if (tN * (NT * 16) >= M) return;

  const int colB = tN * (NT * 16) + (lane & 15);
  const int rowA = tM * 16 + (lane & 15);
  const int aoff = (lane >> 4) * 16;   // A: lanes 16-31 hold K+16..K+31
  const int boff = (lane >> 4) * 16;   // B: lanes 16-31 hold K+16..K+31

  const half_t* wrow = W + (size_t)rowA * Cinp;
  const half_t* xp[NT];
  #pragma unroll
  for (int i = 0; i < NT; ++i)
    xp[i] = X + (size_t)b * xS + (size_t)(colB + 16 * i) * xLd;

  v8f acc[NT];
  #pragma unroll
  for (int i = 0; i < NT; ++i) acc[i] = (v8f){};

  for (int k0 = 0; k0 < Cinp; k0 += 32) {
    if (k0 + 32 < Cinp) {
      #pragma unroll
      for (int i = 0; i < NT; ++i)
        __builtin_prefetch(xp[i] + k0 + 32 + boff, 0, 1);
    }
    v8h al = *(const v8h*)(wrow + k0 + aoff);
    v8h ah = *(const v8h*)(wrow + k0 + aoff + 8);
    v16h a = __builtin_shufflevector(al, ah, 0,1,2,3,4,5,6,7,8,9,10,11,12,13,14,15);
    #pragma unroll
    for (int i = 0; i < NT; ++i) {
      v8h bl = *(const v8h*)(xp[i] + k0 + boff);
      v8h bh = *(const v8h*)(xp[i] + k0 + boff + 8);
      v16h bf = __builtin_shufflevector(bl, bh, 0,1,2,3,4,5,6,7,8,9,10,11,12,13,14,15);
      acc[i] = __builtin_amdgcn_wmma_f32_16x16x32_f16(
          false, a, false, bf, (short)0, acc[i], false, false);
    }
  }

  const float bnmul = 0.99999500003749968f;  // rsqrt(1 + 1e-5)
  const int obase = tM * 16 + ((lane >> 4) << 3);  // C/D: VGPR r -> channel obase+r
  half_t* yb = Y + (size_t)b * yS;

  #pragma unroll
  for (int i = 0; i < NT; ++i) {
    const int col = colB + 16 * i;
    v8h outv;
    #pragma unroll
    for (int r = 0; r < 8; ++r) {
      int o = obase + r;
      float y = acc[i][r];
      if (bias)  y += bias[o];
      if (gamma) y = y * gamma[o] * bnmul + beta[o];
      if (res)   y += (float)res[(size_t)b * resS + (size_t)col * resLd + o];
      if (relu)  y = fmaxf(y, 0.f);
      outv[r] = (half_t)y;
      if (Yf) Yf[(size_t)b * yfS + (size_t)o * M + col] = y;
    }
    *(v8h*)(yb + (size_t)col * yLd + obase) = outv;
  }
}

// ---------------------------------------------------------------------------
// kNN: one thread per query, register insertion-sorted top-16 ascending d2.
// ---------------------------------------------------------------------------
__global__ __launch_bounds__(256) void pt_knn(const float* __restrict__ pts,
                                              int* __restrict__ idx)
{
  int t = blockIdx.x * 256 + threadIdx.x;
  if (t >= BB * NN) return;
  int b = t / NN, n = t % NN;
  const float* px = pts + (size_t)b * 3 * NN;
  const float* py = px + NN;
  const float* pz = py + NN;
  float qx = px[n], qy = py[n], qz = pz[n];
  float bd[16]; int bi[16];
  #pragma unroll
  for (int i = 0; i < 16; ++i) { bd[i] = 3.4e38f; bi[i] = 0; }
  for (int j = 0; j < NN; ++j) {
    float dx = qx - px[j], dy = qy - py[j], dz = qz - pz[j];
    float d = dx * dx + dy * dy + dz * dz;
    if (d < bd[15]) {
      bd[15] = d; bi[15] = j;
      #pragma unroll
      for (int s = 15; s > 0; --s) {
        if (bd[s] < bd[s - 1]) {
          float td = bd[s]; bd[s] = bd[s - 1]; bd[s - 1] = td;
          int   ti = bi[s]; bi[s] = bi[s - 1]; bi[s - 1] = ti;
        }
      }
    }
  }
  int* o = idx + (size_t)t * KK;
  #pragma unroll
  for (int i = 0; i < 16; ++i) o[i] = bi[i];
}

// weights: f32 (Cout,Cin) -> f16 (Cout,Cinp) zero-padded
__global__ void pt_padw(const float* __restrict__ W, half_t* __restrict__ Wh,
                        int Cin, int Cinp, int n)
{
  int t = blockIdx.x * 256 + threadIdx.x;
  if (t >= n) return;
  int o = t / Cinp, kp = t % Cinp;
  Wh[t] = (kp < Cin) ? (half_t)W[o * Cin + kp] : (half_t)0.f;
}

// pts f32 (B,3,N) -> channel-last padded f16 (B,N,32)
__global__ void pt_pad_pts(const float* __restrict__ pts, half_t* __restrict__ o)
{
  int t = blockIdx.x * 256 + threadIdx.x;
  if (t >= BB * NN * 32) return;
  int b = t / (NN * 32), r = t % (NN * 32), n = r / 32, c = r % 32;
  o[t] = (c < 3) ? (half_t)pts[(size_t)b * 3 * NN + c * NN + n] : (half_t)0.f;
}

// rel_geo (B,NK,32) channel-last padded; block-invariant, computed once
__global__ void pt_relgeo(const float* __restrict__ pts, const int* __restrict__ idx,
                          half_t* __restrict__ rel)
{
  int t = blockIdx.x * 256 + threadIdx.x;
  if (t >= BB * NKT * 32) return;
  int b = t / (NKT * 32), r = t % (NKT * 32), nk = r / 32, c = r % 32;
  if (c >= 3) { rel[t] = (half_t)0.f; return; }
  int n = nk / KK;
  int j = idx[(size_t)b * NKT + nk];
  const float* p = pts + (size_t)b * 3 * NN;
  rel[t] = (half_t)(p[c * NN + n] - p[c * NN + j]);
}

// pre[b,nk,c] = q[b,n,c] - k[b,idx,c] + ge[b,nk,c]   (all (.,64) channel-last)
__global__ void pt_pre(const half_t* __restrict__ q, const half_t* __restrict__ k,
                       const half_t* __restrict__ ge, const int* __restrict__ idx,
                       half_t* __restrict__ pre)
{
  long t = (long)blockIdx.x * 256 + threadIdx.x;
  if (t >= (long)BB * NKT * 64) return;
  int b  = (int)(t / ((long)NKT * 64));
  long r = t % ((long)NKT * 64);
  int nk = (int)(r / 64), c = (int)(r % 64);
  int n  = nk / KK;
  int j  = idx[(size_t)b * NKT + nk];
  float v = (float)q[((size_t)b * NN + n) * 64 + c]
          - (float)k[((size_t)b * NN + j) * 64 + c]
          + (float)ge[t];
  pre[t] = (half_t)v;
}

// softmax over k (stride 64 in channel-last (B,NK,64)), in place, f32 math
__global__ void pt_softmax_k(half_t* __restrict__ att)
{
  int t = blockIdx.x * 256 + threadIdx.x;
  if (t >= BB * NN * 64) return;
  int b = t / (NN * 64), r = t % (NN * 64), n = r / 64, c = r % 64;
  half_t* p = att + ((size_t)b * NKT + (size_t)n * KK) * 64 + c;
  float v[KK]; float m = -3.4e38f;
  #pragma unroll
  for (int i = 0; i < KK; ++i) { v[i] = (float)p[(size_t)i * 64]; m = fmaxf(m, v[i]); }
  float s = 0.f;
  #pragma unroll
  for (int i = 0; i < KK; ++i) { v[i] = __expf(v[i] - m); s += v[i]; }
  float inv = 1.f / s;
  #pragma unroll
  for (int i = 0; i < KK; ++i) p[(size_t)i * 64] = (half_t)(v[i] * inv);
}

// agg[b,n,c] = sum_k att*(v[b,idx,c]+ge) + feats[b,n,c]
__global__ void pt_agg(const half_t* __restrict__ att, const half_t* __restrict__ v,
                       const half_t* __restrict__ ge, const int* __restrict__ idx,
                       const half_t* __restrict__ feats, long fS, int fLd,
                       half_t* __restrict__ agg)
{
  int t = blockIdx.x * 256 + threadIdx.x;
  if (t >= BB * NN * 64) return;
  int b = t / (NN * 64), r = t % (NN * 64), n = r / 64, c = r % 64;
  const int* ji = idx + (size_t)b * NKT + (size_t)n * KK;
  size_t base = ((size_t)b * NKT + (size_t)n * KK) * 64 + c;
  const half_t* vb = v + (size_t)b * NN * 64;
  float s = (float)feats[(size_t)b * fS + (size_t)n * fLd + c];
  #pragma unroll
  for (int kk = 0; kk < KK; ++kk)
    s += (float)att[base + (size_t)kk * 64] *
         ((float)vb[(size_t)ji[kk] * 64 + c] + (float)ge[base + (size_t)kk * 64]);
  agg[((size_t)b * NN + n) * 64 + c] = (half_t)s;
}

// block-reduce max over n per (b,c); gf is (B,N,128) channel-last
__global__ void pt_gmax(const half_t* __restrict__ gf, float* __restrict__ gmax)
{
  __shared__ float s[256];
  int bc = blockIdx.x;
  int b = bc / 128, c = bc % 128;
  float m = -3.4e38f;
  for (int n = threadIdx.x; n < NN; n += 256)
    m = fmaxf(m, (float)gf[((size_t)b * NN + n) * 128 + c]);
  s[threadIdx.x] = m;
  __syncthreads();
  for (int st = 128; st > 0; st >>= 1) {
    if ((int)threadIdx.x < st) s[threadIdx.x] = fmaxf(s[threadIdx.x], s[threadIdx.x + st]);
    __syncthreads();
  }
  if (threadIdx.x == 0) gmax[bc] = s[0];
}

// broadcast gmax into channels 256..383 of bigcat (B,N,384)
__global__ void pt_tile(const float* __restrict__ gmax, half_t* __restrict__ bigcat)
{
  int t = blockIdx.x * 256 + threadIdx.x;
  if (t >= BB * NN * 128) return;
  int b = t / (NN * 128), r = t % (NN * 128), n = r / 128, c = r % 128;
  bigcat[((size_t)b * NN + n) * 384 + 256 + c] = (half_t)gmax[b * 128 + c];
}

// ---------------------------------------------------------------------------
static void gemm(hipStream_t st, const half_t* W, int Cout, int Cinp,
                 const half_t* X, long xS, int xLd, int M,
                 half_t* Y, long yS, int yLd,
                 const float* bias, const float* gamma, const float* beta,
                 const half_t* res, long resS, int resLd, int relu,
                 float* Yf, long yfS)
{
  if (M >= 16384) {  // big NK GEMMs: 16x64 tiles, 4 wmma per K-step
    dim3 grid((M / 64 + 7) / 8, Cout / 16, BB);
    pt_gemm_wmma<4><<<grid, 256, 0, st>>>(W, Cinp, X, xS, xLd, M, Y, yS, yLd,
                                          bias, gamma, beta, res, resS, resLd,
                                          relu, Yf, yfS);
  } else {           // N=4096 GEMMs: 16x32 tiles, keep occupancy
    dim3 grid((M / 32 + 7) / 8, Cout / 16, BB);
    pt_gemm_wmma<2><<<grid, 256, 0, st>>>(W, Cinp, X, xS, xLd, M, Y, yS, yLd,
                                          bias, gamma, beta, res, resS, resLd,
                                          relu, Yf, yfS);
  }
}

extern "C" void kernel_launch(void* const* d_in, const int* in_sizes, int n_in,
                              void* d_out, int out_size, void* d_ws, size_t ws_size,
                              hipStream_t stream)
{
  (void)in_sizes; (void)n_in; (void)out_size;
  const float* pts = (const float*)d_in[0];

  auto P  = [&](int i) { return (const float*)d_in[i]; };
  // attn a, field j: qw qb kw kb vw vb g1w g1b g2w g2b r1w r1b rg rb r2w r2b o1w o1b o2w o2b
  auto AP = [&](int a, int j) { return (const float*)d_in[5 + 20 * a + j]; };
  const int CONV0 = 185, GLOB0 = 197, FUSE0 = 201;

  // ---- workspace carve-up --------------------------------------------------
  char* wsp = (char*)d_ws;
  size_t off = 0;
  auto take = [&](size_t bytes) -> void* {
    void* p = wsp + off;
    off = (off + bytes + 255) & ~(size_t)255;
    return p;
  };
  int*    idx    = (int*)   take((size_t)BB * NKT * 4);
  half_t* pts16  = (half_t*)take((size_t)BB * NN  * 32 * 2);
  half_t* rel    = (half_t*)take((size_t)BB * NKT * 32 * 2);
  half_t* b32    = (half_t*)take((size_t)BB * NKT * 32 * 2);   // g1 out / r1 out
  half_t* gebuf  = (half_t*)take((size_t)BB * NKT * 64 * 2);
  half_t* prebuf = (half_t*)take((size_t)BB * NKT * 64 * 2);   // pre / logits / attn
  half_t* q16    = (half_t*)take((size_t)BB * NN * 64 * 2);
  half_t* k16    = (half_t*)take((size_t)BB * NN * 64 * 2);
  half_t* v16    = (half_t*)take((size_t)BB * NN * 64 * 2);
  half_t* agg16  = (half_t*)take((size_t)BB * NN * 64 * 2);
  half_t* t64    = (half_t*)take((size_t)BB * NN * 64 * 2);
  half_t* t128   = (half_t*)take((size_t)BB * NN * 128 * 2);
  half_t* scat   = (half_t*)take((size_t)BB * NN * 192 * 2);
  half_t* bigcat = (half_t*)take((size_t)BB * NN * 384 * 2);
  half_t* fuseh  = (half_t*)take((size_t)BB * NN * 256 * 2);
  float*  gmaxb  = (float*) take((size_t)BB * 128 * 4);

  // f16 padded weight arena (converted once per call)
  auto padw = [&](const float* W, int Cout, int Cin) -> half_t* {
    int Cinp = (Cin + 31) & ~31;
    int n = Cout * Cinp;
    half_t* p = (half_t*)take((size_t)n * 2);
    pt_padw<<<(n + 255) / 256, 256, 0, stream>>>(W, p, Cin, Cinp, n);
    return p;
  };
  half_t* Ws1 = padw(P(1), 64, 3);
  half_t* Ws2 = padw(P(3), 64, 64);
  half_t* Wq[9], *Wk[9], *Wv[9], *Wg1[9], *Wg2[9], *Wr1[9], *Wr2[9], *Wo1[9], *Wo2[9];
  for (int a = 0; a < 9; ++a) {
    Wq[a]  = padw(AP(a, 0), 64, 64);
    Wk[a]  = padw(AP(a, 2), 64, 64);
    Wv[a]  = padw(AP(a, 4), 64, 64);
    Wg1[a] = padw(AP(a, 6), 32, 3);
    Wg2[a] = padw(AP(a, 8), 64, 32);
    Wr1[a] = padw(AP(a, 10), 32, 64);
    Wr2[a] = padw(AP(a, 14), 64, 32);
    Wo1[a] = padw(AP(a, 16), 128, 64);
    Wo2[a] = padw(AP(a, 18), 64, 128);
  }
  half_t* Wc[3];
  for (int s = 0; s < 3; ++s) Wc[s] = padw(P(CONV0 + 4 * s), 64, 192);
  half_t* Wgl1 = padw(P(GLOB0 + 0), 64, 64);
  half_t* Wgl2 = padw(P(GLOB0 + 2), 128, 64);
  half_t* Wf1  = padw(P(FUSE0 + 0), 256, 384);
  half_t* Wf2  = padw(P(FUSE0 + 4), 128, 256);
  if (off > ws_size) return;  // workspace too small

  const long sP = (long)NN * 32, sRel = (long)NKT * 32, s32 = (long)NKT * 32;
  const long sGe = (long)NKT * 64, s64 = (long)NN * 64, s128 = (long)NN * 128;
  const long sScat = (long)NN * 192, sBig = (long)NN * 384, sFu = (long)NN * 256;

  // ---- prologue ------------------------------------------------------------
  pt_knn<<<(BB * NN + 255) / 256, 256, 0, stream>>>(pts, idx);
  pt_pad_pts<<<(BB * NN * 32 + 255) / 256, 256, 0, stream>>>(pts, pts16);
  pt_relgeo<<<(BB * NKT * 32 + 255) / 256, 256, 0, stream>>>(pts, idx, rel);

  // stem -> feat0 at bigcat chan 0
  gemm(stream, Ws1, 64, 32, pts16, sP, 32, NN, t64, s64, 64, P(2), 0, 0, 0, 0, 0, 1, 0, 0);
  gemm(stream, Ws2, 64, 64, t64, s64, 64, NN, bigcat, sBig, 384, P(4), 0, 0, 0, 0, 0, 1, 0, 0);

  // ---- 3 stages x 3 attention blocks --------------------------------------
  for (int si = 0; si < 3; ++si) {
    const half_t* xin = bigcat + (size_t)si * 64;  // feat_{si}, channel-last slice
    long xs = sBig; int xld = 384;
    for (int j = 0; j < 3; ++j) {
      int a = si * 3 + j;
      gemm(stream, Wq[a], 64, 64, xin, xs, xld, NN, q16, s64, 64, AP(a, 1), 0, 0, 0, 0, 0, 0, 0, 0);
      gemm(stream, Wk[a], 64, 64, xin, xs, xld, NN, k16, s64, 64, AP(a, 3), 0, 0, 0, 0, 0, 0, 0, 0);
      gemm(stream, Wv[a], 64, 64, xin, xs, xld, NN, v16, s64, 64, AP(a, 5), 0, 0, 0, 0, 0, 0, 0, 0);
      // ge = conv2(relu(conv1(rel_geo)))
      gemm(stream, Wg1[a], 32, 32, rel, sRel, 32, NKT, b32, s32, 32, AP(a, 7), 0, 0, 0, 0, 0, 1, 0, 0);
      gemm(stream, Wg2[a], 64, 32, b32, s32, 32, NKT, gebuf, sGe, 64, AP(a, 9), 0, 0, 0, 0, 0, 0, 0, 0);
      // pre = q - gather(k) + ge
      pt_pre<<<(int)(((long)BB * NKT * 64 + 255) / 256), 256, 0, stream>>>(q16, k16, gebuf, idx, prebuf);
      // h = relu(bn(conv(pre)));  logits = conv(h) (reuse prebuf)
      gemm(stream, Wr1[a], 32, 64, prebuf, sGe, 64, NKT, b32, s32, 32, AP(a, 11), AP(a, 12), AP(a, 13), 0, 0, 0, 1, 0, 0);
      gemm(stream, Wr2[a], 64, 32, b32, s32, 32, NKT, prebuf, sGe, 64, AP(a, 15), 0, 0, 0, 0, 0, 0, 0, 0);
      pt_softmax_k<<<(BB * NN * 64 + 255) / 256, 256, 0, stream>>>(prebuf);
      // agg = sum_k attn*(gather(v)+ge) + feats
      pt_agg<<<(BB * NN * 64 + 255) / 256, 256, 0, stream>>>(prebuf, v16, gebuf, idx, xin, xs, xld, agg16);
      // out = conv(relu(conv(agg))) + agg -> scat at chan 64*j
      gemm(stream, Wo1[a], 128, 64, agg16, s64, 64, NN, t128, s128, 128, AP(a, 17), 0, 0, 0, 0, 0, 1, 0, 0);
      gemm(stream, Wo2[a], 64, 128, t128, s128, 128, NN, scat + (size_t)j * 64, sScat, 192,
           AP(a, 19), 0, 0, agg16, s64, 64, 0, 0, 0);
      xin = scat + (size_t)j * 64; xs = sScat; xld = 192;
    }
    // stage conv -> feat_{si+1} at bigcat chan 64*(si+1)
    gemm(stream, Wc[si], 64, 192, scat, sScat, 192, NN, bigcat + (size_t)(64 * (si + 1)), sBig, 384,
         P(CONV0 + 4 * si + 1), P(CONV0 + 4 * si + 2), P(CONV0 + 4 * si + 3), 0, 0, 0, 1, 0, 0);
  }

  // ---- global branch -------------------------------------------------------
  gemm(stream, Wgl1, 64, 64, bigcat + 192, sBig, 384, NN, t64, s64, 64, P(GLOB0 + 1), 0, 0, 0, 0, 0, 1, 0, 0);
  gemm(stream, Wgl2, 128, 64, t64, s64, 64, NN, t128, s128, 128, P(GLOB0 + 3), 0, 0, 0, 0, 0, 1, 0, 0);
  pt_gmax<<<BB * 128, 256, 0, stream>>>(t128, gmaxb);
  pt_tile<<<(BB * NN * 128 + 255) / 256, 256, 0, stream>>>(gmaxb, bigcat);

  // ---- fuse; final writes f32 channel-first d_out --------------------------
  gemm(stream, Wf1, 256, 384, bigcat, sBig, 384, NN, fuseh, sFu, 256,
       P(FUSE0 + 1), P(FUSE0 + 2), P(FUSE0 + 3), 0, 0, 0, 1, 0, 0);
  gemm(stream, Wf2, 128, 256, fuseh, sFu, 256, NN, t128, s128, 128,
       P(FUSE0 + 5), P(FUSE0 + 6), P(FUSE0 + 7), 0, 0, 0, 1, (float*)d_out, s128);
}
